// GRU_70789650973283
// MI455X (gfx1250) — matrix operand order repaired and checked by reading, not compiled, not run on hardware
//
#include <hip/hip_runtime.h>

#define T_STEPS 2048
#define BATCH   64
#define D_IN    512
#define D_HID   512
#define KDIM    (D_IN + D_HID)          // 1024
#define NWG     16                      // == max workgroups per cluster
#define NCOL    32                      // output columns per WG (per gate)
#define THREADS 256                     // 8 wave32 -> 8 16x16 output tiles
#define LDS_W_ELEMS (3 * NCOL * KDIM)   // 98304 bf16  (192 KB)
#define LDS_X_ELEMS (BATCH * D_IN)      // 32768 bf16  (64 KB)
#define LDS_H_ELEMS (BATCH * D_HID)     // 32768 bf16  (64 KB)
#define LDS_BYTES   ((LDS_W_ELEMS + LDS_X_ELEMS + LDS_H_ELEMS) * 2) // 327680 = 320 KB

typedef __attribute__((ext_vector_type(8)))  __bf16 v8bf;
typedef __attribute__((ext_vector_type(16))) __bf16 v16bf;
typedef __attribute__((ext_vector_type(8)))  float  v8f;

// A fragment (16x32 bf16, MxK): lane L holds row M=L%16; lanes 0-15 carry
// K = k0..k0+15, lanes 16-31 carry K = k0+16..k0+31.
__device__ __forceinline__ v16bf load_a_frag(const __bf16* p) {
  v8bf lo = *(const v8bf*)(p);        // 8 bf16 @ +0
  v8bf hi = *(const v8bf*)(p + 8);    // 8 bf16 @ +8 elements
  return __builtin_shufflevector(lo, hi, 0, 1, 2, 3, 4, 5, 6, 7,
                                 8, 9, 10, 11, 12, 13, 14, 15);
}

__device__ __forceinline__ v8f wmma_bf16(v16bf a, v16bf b, v8f c) {
  return __builtin_amdgcn_wmma_f32_16x16x32_bf16(false, a, false, b,
                                                 (short)0, c, false, false);
}

// Raw-byte async copy global -> LDS (16B per lane), tracked by ASYNCcnt.
__device__ __forceinline__ void async_b128(unsigned lds_off, const void* gsrc) {
  asm volatile("global_load_async_to_lds_b128 %0, %1, off"
               :: "v"(lds_off), "v"(gsrc) : "memory");
}
__device__ __forceinline__ void wait_async0() {
  asm volatile("s_wait_asynccnt 0" ::: "memory");
}

// Grid-wide barrier: cluster-barrier fast path (HW, 16 WGs on one SE) plus
// agent-scope atomic fallback when not launched as a cluster (ClusterID==0).
__device__ __forceinline__ void grid_barrier(unsigned* bar, unsigned target) {
  __threadfence();
  __syncthreads();
  __builtin_amdgcn_s_cluster_barrier();   // NOP when not in a cluster
  if (__builtin_amdgcn_cluster_id_x() == 0 && threadIdx.x == 0) {
    __hip_atomic_fetch_add(bar, 1u, __ATOMIC_RELEASE, __HIP_MEMORY_SCOPE_AGENT);
    while (__hip_atomic_load(bar, __ATOMIC_ACQUIRE, __HIP_MEMORY_SCOPE_AGENT) < target)
      __builtin_amdgcn_s_sleep(2);
  }
  __syncthreads();
  __threadfence();
}

__global__ void __launch_bounds__(THREADS, 1)
gru_persistent(const float* __restrict__ x,
               const float* __restrict__ Wr, const float* __restrict__ br,
               const float* __restrict__ Wz, const float* __restrict__ bz,
               const float* __restrict__ Wg, const float* __restrict__ bg,
               float* __restrict__ out,
               __bf16* __restrict__ h_buf, __bf16* __restrict__ rh_buf,
               unsigned* __restrict__ bar) {
  extern __shared__ __align__(128) char smem[];
  __bf16* lds_w  = (__bf16*)smem;                                   // [3][NCOL][KDIM]
  __bf16* lds_x  = (__bf16*)(smem + LDS_W_ELEMS * 2);               // [BATCH][D_IN]
  __bf16* lds_hr = (__bf16*)(smem + (LDS_W_ELEMS + LDS_X_ELEMS) * 2); // [BATCH][D_HID]

  const int wg    = blockIdx.x;
  const int tid   = threadIdx.x;
  const int lane  = tid & 31;
  const int wave  = tid >> 5;
  const int mi    = wave & 3;      // M tile (rows 16*mi..16*mi+15)
  const int ni    = wave >> 2;     // N tile within this WG's 32-col slice
  const int khalf = lane >> 4;
  const int nl    = lane & 15;
  const int col_l = ni * 16 + nl;          // 0..31 within slice
  const int gcol  = wg * NCOL + col_l;     // global output column

  // One-time preload: fp32 weights -> bf16 LDS, layout [gate][ncol][K].
  for (int idx = tid; idx < LDS_W_ELEMS; idx += THREADS) {
    int k    = idx & (KDIM - 1);
    int row  = idx >> 10;                  // 0..95
    int gate = row >> 5;
    int ncol = row & (NCOL - 1);
    const float* W = (gate == 0) ? Wr : ((gate == 1) ? Wz : Wg);
    lds_w[idx] = (__bf16)W[(size_t)(wg * NCOL + ncol) * KDIM + k];
  }
  const float brv = br[gcol];
  const float bzv = bz[gcol];
  const float bgv = bg[gcol];
  __syncthreads();

  // B fragments: 32x16 KxN bf16; lane = column nl, K-half selected by khalf;
  // 16 contiguous bf16 (32 B) per lane since W is [N][K] row-major.
  const __bf16* wR = lds_w + (size_t)(0 * NCOL + col_l) * KDIM + khalf * 16;
  const __bf16* wZ = lds_w + (size_t)(1 * NCOL + col_l) * KDIM + khalf * 16;
  const __bf16* wG = lds_w + (size_t)(2 * NCOL + col_l) * KDIM + khalf * 16;

  const int rowA = mi * 16 + nl;
  const __bf16* aX = lds_x  + rowA * D_IN  + khalf * 16;
  const __bf16* aH = lds_hr + rowA * D_HID + khalf * 16;

  const unsigned lds_hr_off = (unsigned)(size_t)(const void*)lds_hr;

  unsigned barphase = 0;

  for (int t = 0; t < T_STEPS; ++t) {
    // Stage 1: async raw-copy h (bf16) into LDS, overlap with x_t cvt fp32->bf16.
    for (int c = tid; c < LDS_H_ELEMS / 8; c += THREADS)
      async_b128(lds_hr_off + (unsigned)c * 16u, (const char*)h_buf + (size_t)c * 16);
    {
      const float4* xs = (const float4*)(x + (size_t)t * (BATCH * D_IN));
      for (int c = tid; c < (BATCH * D_IN) / 4; c += THREADS) {
        float4 v = xs[c];
        __bf16* d = lds_x + c * 4;
        d[0] = (__bf16)v.x; d[1] = (__bf16)v.y;
        d[2] = (__bf16)v.z; d[3] = (__bf16)v.w;
      }
    }
    wait_async0();
    __syncthreads();

    // Phase A: r and z slices (fused K-loop reusing the A fragment).
    v8f accR = {};
    v8f accZ = {};
    for (int kk = 0; kk < D_IN; kk += 32) {
      v16bf a = load_a_frag(aX + kk);
      accR = wmma_bf16(a, *(const v16bf*)(wR + kk), accR);
      accZ = wmma_bf16(a, *(const v16bf*)(wZ + kk), accZ);
    }
    for (int kk = 0; kk < D_HID; kk += 32) {
      v16bf a = load_a_frag(aH + kk);
      accR = wmma_bf16(a, *(const v16bf*)(wR + D_IN + kk), accR);
      accZ = wmma_bf16(a, *(const v16bf*)(wZ + D_IN + kk), accZ);
    }

    // C/D layout: VGPR i, lanes 0-15 -> M=i, lanes 16-31 -> M=8+i; N = lane&15.
    float hold[8], zv[8];
#pragma unroll
    for (int i = 0; i < 8; ++i) {
      int row  = mi * 16 + khalf * 8 + i;
      float hv = (float)lds_hr[row * D_HID + gcol];
      float rv = 1.0f / (1.0f + __expf(-(accR[i] + brv)));
      float z  = 1.0f / (1.0f + __expf(-(accZ[i] + bzv)));
      hold[i] = hv;
      zv[i]   = z;
      rh_buf[row * D_HID + gcol] = (__bf16)(rv * hv);   // share r*h slice
    }

    grid_barrier(bar, ++barphase * NWG);

    // Pull the full r*h vector into LDS (async raw copy, overwrites h region).
    for (int c = tid; c < LDS_H_ELEMS / 8; c += THREADS)
      async_b128(lds_hr_off + (unsigned)c * 16u, (const char*)rh_buf + (size_t)c * 16);
    wait_async0();
    __syncthreads();

    // Phase B: candidate g, then h_new = z*h + (1-z)*g.
    v8f accG = {};
    for (int kk = 0; kk < D_IN; kk += 32)
      accG = wmma_bf16(load_a_frag(aX + kk), *(const v16bf*)(wG + kk), accG);
    for (int kk = 0; kk < D_HID; kk += 32)
      accG = wmma_bf16(load_a_frag(aH + kk), *(const v16bf*)(wG + D_IN + kk), accG);

    float* out_t = out + (size_t)t * (BATCH * D_HID);
#pragma unroll
    for (int i = 0; i < 8; ++i) {
      int row  = mi * 16 + khalf * 8 + i;
      float e  = __expf(2.0f * (accG[i] + bgv));        // tanh via exp
      float g  = (e - 1.0f) / (e + 1.0f);
      float hn = zv[i] * hold[i] + (1.0f - zv[i]) * g;
      out_t[row * D_HID + gcol] = hn;
      h_buf[row * D_HID + gcol] = (__bf16)hn;
      if (t == T_STEPS - 1)
        out[(size_t)T_STEPS * BATCH * D_HID + row * D_HID + gcol] = hn;
    }

    grid_barrier(bar, ++barphase * NWG);
  }
}

extern "C" void kernel_launch(void* const* d_in, const int* in_sizes, int n_in,
                              void* d_out, int out_size, void* d_ws, size_t ws_size,
                              hipStream_t stream) {
  const float* x  = (const float*)d_in[0];
  const float* Wr = (const float*)d_in[1];
  const float* br = (const float*)d_in[2];
  const float* Wz = (const float*)d_in[3];
  const float* bz = (const float*)d_in[4];
  const float* Wg = (const float*)d_in[5];
  const float* bg = (const float*)d_in[6];
  float* out = (float*)d_out;

  char* ws = (char*)d_ws;
  unsigned* bar  = (unsigned*)ws;                              // 256 B
  __bf16* h_buf  = (__bf16*)(ws + 256);                        // 64 KB
  __bf16* rh_buf = (__bf16*)(ws + 256 + BATCH * D_HID * 2);    // 64 KB

  // Zero barrier counter + initial hidden state every launch (graph-safe).
  hipMemsetAsync(ws, 0, 256 + BATCH * D_HID * 2, stream);

  hipFuncSetAttribute(reinterpret_cast<const void*>(gru_persistent),
                      hipFuncAttributeMaxDynamicSharedMemorySize, LDS_BYTES);

  gru_persistent<<<dim3(NWG), dim3(THREADS), LDS_BYTES, stream>>>(
      x, Wr, br, Wz, bz, Wg, bg, out, h_buf, rh_buf, bar);

  (void)in_sizes; (void)n_in; (void)out_size; (void)ws_size;
}
